// SynthesisLayer_18786186953338
// MI455X (gfx1250) — matrix operand order repaired and checked by reading, not compiled, not run on hardware
//
#include <hip/hip_runtime.h>

#define IN_F 512
#define OUT_F 512
#define LAT 512
#define RES 64
#define BATCH 8
#define EPS 1e-8f

typedef __bf16 bf16_t;
typedef __attribute__((ext_vector_type(16))) __bf16 v16bf;
typedef __attribute__((ext_vector_type(4)))  __bf16 v4bf;
typedef __attribute__((ext_vector_type(8)))  float  v8f;

// C_LIN = 1/sqrt(512), C_CONV = 1/sqrt(512*9)
#define C_LIN_F  0.04419417382415922f
#define C_CONV_F 0.014731391274719736f

// s[b,i] = sum_l latent[b,l] * w_lin[i,l] * C_LIN + b_lin[i]
__global__ void k_style(const float* __restrict__ latent, const float* __restrict__ w_lin,
                        const float* __restrict__ b_lin, float* __restrict__ s) {
  int tid = blockIdx.x * blockDim.x + threadIdx.x;   // [0, B*IN_F)
  int b = tid >> 9, i = tid & 511;
  const float* lp = latent + b * LAT;
  const float* wp = w_lin + i * LAT;
  float acc = 0.f;
  for (int l = 0; l < LAT; ++l) acc += lp[l] * wp[l];
  s[tid] = acc * C_LIN_F + b_lin[i];
}

// w2[o,i] = C_CONV^2 * sum_kk w_conv[o,i,kk]^2
__global__ void k_w2(const float* __restrict__ w_conv, float* __restrict__ w2) {
  int tid = blockIdx.x * blockDim.x + threadIdx.x;   // [0, OUT_F*IN_F)
  const float* wp = w_conv + tid * 9;
  float acc = 0.f;
  for (int t = 0; t < 9; ++t) { float w = wp[t]; acc += w * w; }
  w2[tid] = acc * (C_CONV_F * C_CONV_F);
}

// sigma[b,o] = rsqrt( sum_i w2[o,i] * s[b,i]^2 + eps )
__global__ void k_sigma(const float* __restrict__ w2, const float* __restrict__ s,
                        float* __restrict__ sigma) {
  int tid = blockIdx.x * blockDim.x + threadIdx.x;   // [0, B*OUT_F)
  int b = tid >> 9, o = tid & 511;
  const float* wp = w2 + o * IN_F;
  const float* sp = s + b * IN_F;
  float acc = 0.f;
  for (int i = 0; i < IN_F; ++i) { float sv = sp[i]; acc += wp[i] * sv * sv; }
  sigma[tid] = rsqrtf(acc + EPS);
}

// Pre-swizzle weights into WMMA A-fragment order:
// wpack[mt][tap][kc][lane][j], 512 bf16 per fragment.
// A 16x32 layout (contiguous K, matching B): lane<16 holds row M=lane, K 0-15 (j 0-15);
// lane>=16 holds row M=lane-16, K 16-31 (j 0-15).
__global__ void k_pack(const float* __restrict__ w_conv, bf16_t* __restrict__ wpack) {
  int idx = blockIdx.x * blockDim.x + threadIdx.x;   // [0, OUT_F*IN_F*9)
  int e    = idx & 511;
  int frag = idx >> 9;
  int lane = e >> 4;
  int j    = e & 15;
  int kc   = frag & 15;
  int t2   = frag >> 4;
  int tap  = t2 % 9;
  int mt   = t2 / 9;
  int m = mt * 16 + (lane & 15);
  int k;
  if (lane < 16) k = j;
  else           k = j + 16;
  int i = kc * 32 + k;
  wpack[idx] = (bf16_t)(w_conv[(m * IN_F + i) * 9 + tap] * C_CONV_F);
}

// xb = bf16(x * s[b,i])
__global__ void k_xmod(const float* __restrict__ x, const float* __restrict__ s,
                       bf16_t* __restrict__ xb) {
  int idx = blockIdx.x * blockDim.x + threadIdx.x;   // [0, B*IN_F*RES*RES)
  int bi = idx >> 12;                                // b*IN_F + i
  xb[idx] = (bf16_t)(x[idx] * s[bi]);
}

#define XS_ELEMS (3 * 66 * 32)   // one buffer: [row 3][col+1 66][chan 32]

// Load next 32-channel chunk (3 rows, 64 cols) as coalesced 8B quads into regs.
__device__ __forceinline__ void load_chunk(const bf16_t* __restrict__ xb, int b, int h,
                                           int kc, int tid, v4bf rv[6]) {
#pragma unroll
  for (int t = 0; t < 6; ++t) {
    int q    = tid + t * 256;         // [0, 1536) quads
    int col4 = (q & 15) * 4;
    int r    = (q >> 4) % 3;
    int ch   = q / 48;
    int row  = h - 1 + r;
    v4bf v = {};
    if (row >= 0 && row < RES)
      v = *(const v4bf*)(xb + ((b * IN_F + kc * 32 + ch) * RES + row) * RES + col4);
    rv[t] = v;
  }
}

// Commit staged quads to LDS, transposed to [row][col+1][chan].
__device__ __forceinline__ void store_chunk(bf16_t* buf, int tid, const v4bf rv[6]) {
#pragma unroll
  for (int t = 0; t < 6; ++t) {
    int q    = tid + t * 256;
    int col4 = (q & 15) * 4;
    int r    = (q >> 4) % 3;
    int ch   = q / 48;
#pragma unroll
    for (int j = 0; j < 4; ++j)
      buf[r * 2112 + (col4 + 1 + j) * 32 + ch] = rv[t][j];
  }
}

// Implicit-GEMM 3x3 conv. Block (mblock, h, b): y[b, 256 outs, h, 0..63].
// 8 waves, each 32M x 64N (2 A-frags, 4 B-frags, 8 WMMA per tap).
// K-loop: 16 chunks of 32 input channels, double-buffered LDS staging.
__global__ __launch_bounds__(256) void k_conv(
    const bf16_t* __restrict__ xb, const bf16_t* __restrict__ wpack,
    const float* __restrict__ sigma, const float* __restrict__ b_conv,
    float* __restrict__ y) {
  __shared__ __align__(32) bf16_t xs[2 * XS_ELEMS];  // 24.8 KB double buffer

  const int tid    = threadIdx.x;
  const int lane   = tid & 31;
  const int wm     = tid >> 5;      // 0..7 : 32-row M sub-block
  const int mblock = blockIdx.x;    // 0..1
  const int h      = blockIdx.y;    // 0..63
  const int b      = blockIdx.z;    // 0..7

  // zero halo columns (col 0 and 65) of both buffers, once
  if (tid < 192) {
    int r = tid / 64, rem = tid % 64;
    int c = rem >> 5, ch = rem & 31;
    int off = r * 2112 + (c ? 65 * 32 : 0) + ch;
    xs[off]            = (bf16_t)0.0f;
    xs[XS_ELEMS + off] = (bf16_t)0.0f;
  }

  v8f acc[2][4] = {};

  const int khalf = (lane < 16) ? 0 : 16;  // B-frag: lanes 0-15 K0-15, lanes 16-31 K16-31
  const int ln    = lane & 15;
  const int mt0   = mblock * 16 + wm * 2;

  v4bf rv[6];
  load_chunk(xb, b, h, 0, tid, rv);
  store_chunk(xs, tid, rv);
  __syncthreads();

  for (int kc = 0; kc < 16; ++kc) {
    const bf16_t* cur = xs + (kc & 1) * XS_ELEMS;
    if (kc + 1 < 16) load_chunk(xb, b, h, kc + 1, tid, rv);  // hide behind WMMAs

#pragma unroll
    for (int dy = 0; dy < 3; ++dy) {
#pragma unroll
      for (int dx = 0; dx < 3; ++dx) {
        const int tap = dy * 3 + dx;
        // A fragments: one coalesced 32B read per lane from pre-swizzled weights (L2)
        v16bf a0 = *(const v16bf*)(wpack + (((mt0    ) * 9 + tap) * 16 + kc) * 512 + lane * 16);
        v16bf a1 = *(const v16bf*)(wpack + (((mt0 + 1) * 9 + tap) * 16 + kc) * 512 + lane * 16);
        // Hoist all 4 B-fragment LDS loads into distinct registers so the
        // scheduler can issue the 8 ds_load_b128 up front and drain them with
        // partial dscnt waits while earlier WMMAs execute.
        v16bf bfr[4];
#pragma unroll
        for (int ni = 0; ni < 4; ++ni) {
          const int cb = ni * 16 + ln + dx;    // halo shift = +dx column offset in LDS
          bfr[ni] = *(const v16bf*)&cur[dy * 2112 + cb * 32 + khalf];
        }
#pragma unroll
        for (int ni = 0; ni < 4; ++ni) {
          acc[0][ni] = __builtin_amdgcn_wmma_f32_16x16x32_bf16(false, a0, false, bfr[ni], (short)0, acc[0][ni], false, false);
          acc[1][ni] = __builtin_amdgcn_wmma_f32_16x16x32_bf16(false, a1, false, bfr[ni], (short)0, acc[1][ni], false, false);
        }
      }
    }

    if (kc + 1 < 16) store_chunk(xs + ((kc + 1) & 1) * XS_ELEMS, tid, rv);
    __syncthreads();
  }

  // epilogue: C/D layout — VGPR v: lanes 0-15 -> M=v, lanes 16-31 -> M=v+8; N = lane&15
  const int m_base   = mblock * 256 + wm * 32;
  const int row_half = (lane < 16) ? 0 : 8;
#pragma unroll
  for (int mi = 0; mi < 2; ++mi) {
#pragma unroll
    for (int v = 0; v < 8; ++v) {
      int o = m_base + mi * 16 + row_half + v;
      float sg = sigma[b * OUT_F + o];
      float bc = b_conv[o];
      float* yp = y + ((b * OUT_F + o) * RES + h) * RES;
#pragma unroll
      for (int ni = 0; ni < 4; ++ni)
        yp[ni * 16 + ln] = acc[mi][ni][v] * sg + bc;
    }
  }
}

extern "C" void kernel_launch(void* const* d_in, const int* in_sizes, int n_in,
                              void* d_out, int out_size, void* d_ws, size_t ws_size,
                              hipStream_t stream) {
  const float* x      = (const float*)d_in[0];
  const float* latent = (const float*)d_in[1];
  const float* w_lin  = (const float*)d_in[2];
  const float* b_lin  = (const float*)d_in[3];
  const float* w_conv = (const float*)d_in[4];
  const float* b_conv = (const float*)d_in[5];
  float* y = (float*)d_out;

  char* ws = (char*)d_ws;
  float*  s     = (float*)(ws);                                   // 16 KB
  float*  w2    = (float*)(ws + (16 << 10));                      // 1 MB
  float*  sigma = (float*)(ws + (16 << 10) + (1 << 20));          // 16 KB
  bf16_t* wpack = (bf16_t*)(ws + (32 << 10) + (1 << 20));         // 4.5 MB
  bf16_t* xb    = (bf16_t*)(ws + (32 << 10) + (1 << 20) + OUT_F * IN_F * 9 * 2); // 32 MB

  k_style<<<(BATCH * IN_F) / 256, 256, 0, stream>>>(latent, w_lin, b_lin, s);
  k_w2<<<(OUT_F * IN_F) / 256, 256, 0, stream>>>(w_conv, w2);
  k_sigma<<<(BATCH * OUT_F) / 256, 256, 0, stream>>>(w2, s, sigma);
  k_pack<<<(OUT_F * IN_F * 9) / 256, 256, 0, stream>>>(w_conv, wpack);
  k_xmod<<<(BATCH * IN_F * RES * RES) / 256, 256, 0, stream>>>(x, s, xb);

  dim3 grid(2, RES, BATCH);
  k_conv<<<grid, 256, 0, stream>>>(xb, wpack, sigma, b_conv, y);
}
